// ActivationUnit_47021301957261
// MI455X (gfx1250) — matrix operand order repaired and checked by reading, not compiled, not run on hardware
//
#include <hip/hip_runtime.h>

#define B_ 256
#define F_ 8
#define S_ 200
#define SP_ 208          // padded S for the h workspace
#define D_ 64
#define H_ 36
#define CNT_ (B_ * S_)   // 51200 elements per (f,h) statistic

typedef __bf16 bf4   __attribute__((ext_vector_type(4)));
typedef __bf16 bf8   __attribute__((ext_vector_type(8)));
typedef __bf16 v16bf __attribute__((ext_vector_type(16)));
typedef float  f4    __attribute__((ext_vector_type(4)));
typedef float  v8f   __attribute__((ext_vector_type(8)));

// A-fragment: two 8-half runs, second at +16 halfs from kbase.
// Call with base already offset by (kbase + 8*g).
static __device__ inline v16bf ld_pair8(const __bf16* base) {
  bf8 lo = *(const bf8*)(base);
  bf8 hi = *(const bf8*)(base + 16);
  return __builtin_shufflevector(lo, hi, 0,1,2,3,4,5,6,7,8,9,10,11,12,13,14,15);
}

// 16 contiguous halfs; unused (B-fragments use ld_pair8).
static __device__ inline v16bf ld_run16(const __bf16* base) {
  bf8 lo = *(const bf8*)(base);
  bf8 hi = *(const bf8*)(base + 8);
  return __builtin_shufflevector(lo, hi, 0,1,2,3,4,5,6,7,8,9,10,11,12,13,14,15);
}

static __device__ inline v16bf bsub(v16bf x, v16bf y) {
  v16bf r;
#pragma unroll
  for (int i = 0; i < 16; ++i) r[i] = (__bf16)((float)x[i] - (float)y[i]);
  return r;
}

// ---------------------------------------------------------------------------
// Phase 1: h = cat(key,q,q-key) @ W_h + b_h  via bf16 WMMA; partial stats.
// One workgroup per (b,f). 256 threads = 8 waves.
// h workspace layout: [p][h][s] with s padded to 208 -> vector stores and
// fully coalesced phase-3 reads.
// ---------------------------------------------------------------------------
__global__ __launch_bounds__(256) void au_phase1(
    const float* __restrict__ query, const float* __restrict__ key,
    const float* __restrict__ W_h, const float* __restrict__ b_h,
    float* __restrict__ hws, float* __restrict__ gsum, float* __restrict__ gsq)
{
  // key tile: 208 rows (pad 8 zero rows) x 64 cols, row stride 72 halfs (bank-safe)
  __shared__ __bf16 ks[208 * 72];
  // W_h transposed: 48 rows (n, pad 36..47 zero) x 192 cols (k), stride 200 halfs
  __shared__ __bf16 wts[48 * 200];
  __shared__ __bf16 qs[64];
  __shared__ float bhs[48];
  __shared__ float ssum[48];
  __shared__ float ssq[48];

  const int p   = blockIdx.x;        // b*F + f
  const int f   = p & (F_ - 1);
  const int tid = threadIdx.x;

  const float* qg = query + (size_t)p * D_;
  const float* kg = key   + (size_t)p * S_ * D_;
  const float* wg = W_h   + (size_t)f * (3 * D_) * H_;

  // key -> LDS bf16, float4 loads + 8B LDS stores
  for (int i = tid; i < 208 * 16; i += 256) {
    int s = i >> 4, d4 = (i & 15) << 2;
    bf4 w;
    if (s < S_) {
      f4 v = *(const f4*)(kg + s * D_ + d4);
      w[0] = (__bf16)v[0]; w[1] = (__bf16)v[1];
      w[2] = (__bf16)v[2]; w[3] = (__bf16)v[3];
    } else {
      w[0] = (__bf16)0.f; w[1] = (__bf16)0.f;
      w[2] = (__bf16)0.f; w[3] = (__bf16)0.f;
    }
    *(bf4*)(ks + s * 72 + d4) = w;
  }
  if (tid < 64) qs[tid] = (__bf16)qg[tid];
  for (int i = tid; i < 192 * 36; i += 256) {       // coalesced on global read
    int k = i / 36, n = i % 36;
    wts[n * 200 + k] = (__bf16)wg[i];
  }
  for (int i = tid; i < 12 * 192; i += 256) {       // zero-pad n = 36..47
    int n = 36 + i / 192, k = i % 192;
    wts[n * 200 + k] = (__bf16)0.0f;
  }
  if (tid < 48) {
    bhs[tid]  = (tid < H_) ? b_h[f * H_ + tid] : 0.0f;
    ssum[tid] = 0.0f;
    ssq[tid]  = 0.0f;
  }
  __syncthreads();

  const int wave = tid >> 5, L = tid & 31, g = L >> 4, r = L & 15;

  float lsum[3] = {0.f, 0.f, 0.f};
  float lsq [3] = {0.f, 0.f, 0.f};
  float bv[3];
#pragma unroll
  for (int nt = 0; nt < 3; ++nt) bv[nt] = bhs[nt * 16 + r];

  // q fragments are row-invariant (broadcast reads from LDS)
  const v16bf q0 = ld_pair8(qs + 8 * g);
  const v16bf q1 = ld_pair8(qs + 32 + 8 * g);

  for (int mt = wave; mt < 13; mt += 8) {           // 13 M-tiles of 16 rows
    const __bf16* krow = ks + (mt * 16 + r) * 72;
    v16bf A[6];
    A[0] = ld_pair8(krow + 8 * g);                  // cat block 0: key[:,0:32]
    A[1] = ld_pair8(krow + 32 + 8 * g);             // key[:,32:64]
    A[2] = q0;                                      // cat block 1: q
    A[3] = q1;
    A[4] = bsub(q0, A[0]);                          // cat block 2: q - key
    A[5] = bsub(q1, A[1]);

    const int srow0 = mt * 16 + 8 * g;              // lane's first output row

#pragma unroll
    for (int nt = 0; nt < 3; ++nt) {                // 3 N-tiles (48 padded cols)
      v8f acc = {0.f, 0.f, 0.f, 0.f, 0.f, 0.f, 0.f, 0.f};
      const __bf16* wrow = wts + (nt * 16 + r) * 200 + 8 * g;
#pragma unroll
      for (int kk = 0; kk < 6; ++kk) {              // K = 192 = 6 x 32
        v16bf bf = ld_pair8(wrow + kk * 32);
        acc = __builtin_amdgcn_wmma_f32_16x16x32_bf16(
            false, A[kk], false, bf, (short)0, acc, false, false);
      }
      const int hcol = nt * 16 + r;                 // output column (h index)
      if (hcol < H_) {
        float* dst = hws + ((size_t)p * H_ + hcol) * SP_ + srow0;
        f4 s0, s1;
#pragma unroll
        for (int i = 0; i < 8; ++i) {
          float v = acc[i] + bv[nt];
          float m = ((srow0 + i) < S_) ? 1.0f : 0.0f;   // branch-free row mask
          lsum[nt] += v * m;
          lsq [nt] += v * v * m;
          if (i < 4) s0[i] = v; else s1[i - 4] = v;
        }
        *(f4*)(dst)     = s0;                       // two 16B stores, 8 consecutive s
        *(f4*)(dst + 4) = s1;
      }
    }
  }

#pragma unroll
  for (int nt = 0; nt < 3; ++nt) {
    int hcol = nt * 16 + r;
    if (hcol < H_) {
      atomicAdd(&ssum[hcol], lsum[nt]);
      atomicAdd(&ssq [hcol], lsq [nt]);
    }
  }
  __syncthreads();
  if (tid < H_) {
    atomicAdd(&gsum[f * H_ + tid], ssum[tid]);
    atomicAdd(&gsq [f * H_ + tid], ssq [tid]);
  }
}

// ---------------------------------------------------------------------------
// Phase 2: finalize mean / rsqrt(var + eps) for F*H = 288 pairs.
// ---------------------------------------------------------------------------
__global__ void au_stats(const float* __restrict__ gsum,
                         const float* __restrict__ gsq,
                         float* __restrict__ mean, float* __restrict__ inv)
{
  int i = blockIdx.x * 64 + threadIdx.x;
  if (i < F_ * H_) {
    float m = gsum[i] * (1.0f / (float)CNT_);
    float v = gsq[i] * (1.0f / (float)CNT_) - m * m;
    mean[i] = m;
    inv[i]  = rsqrtf(v + 0.001f);
  }
}

// ---------------------------------------------------------------------------
// Phase 3: gate + score + masked softmax + attn-weighted key sum.
// One workgroup per (b,f). h reads are contiguous across threads per h.
// ---------------------------------------------------------------------------
__global__ __launch_bounds__(256) void au_phase3(
    const float* __restrict__ key, const float* __restrict__ hws,
    const float* __restrict__ mean, const float* __restrict__ inv,
    const float* __restrict__ alpha, const float* __restrict__ W_o,
    const float* __restrict__ b_o, const int* __restrict__ seqn,
    float* __restrict__ out)
{
  __shared__ float sm[H_], si[H_], sw[H_];
  __shared__ float sattn[256];
  __shared__ float rbuf[256];

  const int p   = blockIdx.x;
  const int f   = p & (F_ - 1);
  const int tid = threadIdx.x;

  if (tid < H_) {
    sm[tid] = mean[f * H_ + tid];
    si[tid] = inv [f * H_ + tid];
    sw[tid] = W_o [f * H_ + tid];
  }
  __syncthreads();

  const float af  = alpha[f];
  const float bo  = b_o[f];
  const int   seq = seqn[p];

  float sval = -__builtin_inff();
  if (tid < S_ && tid < seq) {
    const float* hbase = hws + (size_t)p * H_ * SP_ + tid;
    float accs = bo;
#pragma unroll 4
    for (int h = 0; h < H_; ++h) {
      float hv = hbase[(size_t)h * SP_];          // coalesced across threads
      float z  = (hv - sm[h]) * si[h];
      float pp = 1.0f / (1.0f + __expf(-z));
      accs += (af + pp * (1.0f - af)) * hv * sw[h];
    }
    sval = accs;
  }

  rbuf[tid] = sval;
  __syncthreads();
  for (int st = 128; st > 0; st >>= 1) {
    if (tid < st) rbuf[tid] = fmaxf(rbuf[tid], rbuf[tid + st]);
    __syncthreads();
  }
  const float mx = rbuf[0];
  __syncthreads();

  const float e = __expf(sval - mx);     // exp(-inf) = 0 for masked lanes
  rbuf[tid] = e;
  __syncthreads();
  for (int st = 128; st > 0; st >>= 1) {
    if (tid < st) rbuf[tid] += rbuf[tid + st];
    __syncthreads();
  }
  const float denom = rbuf[0];
  __syncthreads();

  sattn[tid] = e / denom;
  __syncthreads();

  // attn-weighted key sum: 4 chunks of 50 positions x 64 dims
  const int d = tid & 63, c = tid >> 6;
  const float* kb = key + (size_t)p * S_ * D_ + d;
  float acc = 0.f;
  for (int s = c * 50; s < c * 50 + 50; ++s)
    acc += sattn[s] * kb[(size_t)s * D_];
  rbuf[tid] = acc;
  __syncthreads();
  if (tid < 64)
    out[(size_t)p * D_ + tid] =
        rbuf[tid] + rbuf[tid + 64] + rbuf[tid + 128] + rbuf[tid + 192];
}

// ---------------------------------------------------------------------------
extern "C" void kernel_launch(void* const* d_in, const int* in_sizes, int n_in,
                              void* d_out, int out_size, void* d_ws, size_t ws_size,
                              hipStream_t stream)
{
  (void)in_sizes; (void)n_in; (void)out_size; (void)ws_size;
  const float* query = (const float*)d_in[0];
  const float* key   = (const float*)d_in[1];
  const float* W_h   = (const float*)d_in[2];
  const float* b_h   = (const float*)d_in[3];
  const float* alpha = (const float*)d_in[4];
  const float* W_o   = (const float*)d_in[5];
  const float* b_o   = (const float*)d_in[6];
  const int*   seqn  = (const int*)d_in[7];
  float* out = (float*)d_out;

  char*  ws     = (char*)d_ws;
  float* hws    = (float*)ws;                             // B*F * H * 208 f32
  size_t hbytes = (size_t)B_ * F_ * H_ * SP_ * sizeof(float);
  float* gsum   = (float*)(ws + hbytes);
  float* gsq    = gsum + F_ * H_;
  float* meanb  = gsq + F_ * H_;
  float* invb   = meanb + F_ * H_;

  hipMemsetAsync(gsum, 0, 2 * F_ * H_ * sizeof(float), stream);

  au_phase1<<<B_ * F_, 256, 0, stream>>>(query, key, W_h, b_h, hws, gsum, gsq);
  au_stats<<<(F_ * H_ + 63) / 64, 64, 0, stream>>>(gsum, gsq, meanb, invb);
  au_phase3<<<B_ * F_, 256, 0, stream>>>(key, hws, meanb, invb, alpha, W_o, b_o,
                                         seqn, out);
}
